// ImportanceMap_49143015800950
// MI455X (gfx1250) — compile-verified, matrix operand order checked by reading, not run on hardware
//
#include <hip/hip_runtime.h>
#include <hip/hip_bf16.h>

// ---------------- problem constants ----------------
#define BATCH   32
#define NPB     (1024 * 1024)          // elements per batch
#define NBINS   4096
#define XMIN    (-6.0f)
#define BINW    (12.0f / (float)NBINS)
#define INVW    ((float)NBINS / 12.0f)

typedef __attribute__((ext_vector_type(4)))  float    f32x4;
typedef __attribute__((ext_vector_type(4)))  int      i32x4;
typedef __attribute__((ext_vector_type(16))) _Float16 v16h;
typedef __attribute__((ext_vector_type(8)))  float    v8f;
typedef _Float16 f16_t;

typedef __attribute__((address_space(1))) i32x4 as1_i32x4;
typedef __attribute__((address_space(3))) i32x4 as3_i32x4;

// CDNA5 async global->LDS path (guarded; fallback = direct loads)
#if __has_builtin(__builtin_amdgcn_global_load_async_to_lds_b128) && \
    __has_builtin(__builtin_amdgcn_s_wait_asynccnt)
#define HAS_ASYNC_LDS 1
#endif

// Address-space casts via integers (always compile; correct per AMDGPU
// address map: global AS1 shares the 64-bit generic address; LDS AS3 is the
// low 32 bits of the generic LDS address per ISA 10.2 aperture rules).
__device__ __forceinline__ as1_i32x4* to_global_v4(const void* p) {
    return (as1_i32x4*)(unsigned long long)p;
}
__device__ __forceinline__ as3_i32x4* to_lds_v4(void* p) {
    return (as3_i32x4*)(unsigned int)(unsigned long long)p;
}

__device__ __forceinline__ float fast_sigmoid(float t) {
    // 1/(1+exp(-t)) via hardware exp2 + rcp
    float e = __builtin_amdgcn_exp2f(t * -1.44269504088896340736f);
    return __builtin_amdgcn_rcpf(1.0f + e);
}

// ---------------- 0) zero workspace histograms ----------------
__global__ __launch_bounds__(256) void zero_kernel(float* p, int n) {
    int i = blockIdx.x * 256 + threadIdx.x;
    if (i < n) p[i] = 0.0f;
}

// ---------------- 1) per-batch histogram (count + residual) ----------------
// grid = (32, BATCH); each block handles 32768 contiguous elements of one batch.
__global__ __launch_bounds__(256) void hist_kernel(const float* __restrict__ X,
                                                   float* __restrict__ hcnt,
                                                   float* __restrict__ hres) {
    __shared__ float sc[NBINS];
    __shared__ float sr[NBINS];
    for (int i = threadIdx.x; i < NBINS; i += 256) { sc[i] = 0.0f; sr[i] = 0.0f; }
    __syncthreads();

    const int b = blockIdx.y;
    const f32x4* src = (const f32x4*)(X + (size_t)b * NPB) +
                       (size_t)blockIdx.x * (NPB / 32 / 4);
#pragma unroll 4
    for (int it = 0; it < 32; ++it) {
        f32x4 v = src[it * 256 + threadIdx.x];
#pragma unroll
        for (int k = 0; k < 4; ++k) {
            float x  = v[k];
            float xc = fminf(fmaxf(x, -5.99990f), 5.99990f);
            int idx  = (int)fmaf(xc, INVW, (float)(NBINS / 2));
            idx      = max(0, min(NBINS - 1, idx));
            float ct = fmaf((float)idx + 0.5f, BINW, XMIN);
            atomicAdd(&sc[idx], 1.0f);       // ds_add_f32
            atomicAdd(&sr[idx], xc - ct);    // ds_add_f32
        }
    }
    __syncthreads();

    float* gc = hcnt + (size_t)b * NBINS;
    float* gr = hres + (size_t)b * NBINS;
    for (int i = threadIdx.x; i < NBINS; i += 256) {
        float c = sc[i];
        if (c != 0.0f) {
            atomicAdd(&gc[i], c);
            atomicAdd(&gr[i], sr[i]);
        }
    }
}

// ---------------- 2) bisection on histogram, WMMA weighted reduction -------
// grid = BATCH blocks x 256 threads (8 waves); wave w owns bins [512w, 512w+512).
// Wave layout matches 16-bit A (16x32) striping: lane L<16 -> row L, K in
// {0..7}U{16..23}; lane L+16 -> row L, K in {8..15}U{24..31}. B mirrors it so
// the diagonal of D = A*B is the per-chunk dot(count, sigmoid).
__global__ __launch_bounds__(256) void bisect_kernel(const float* __restrict__ hcnt,
                                                     const float* __restrict__ hres,
                                                     float* __restrict__ cvals,
                                                     float* __restrict__ out_tail) {
    const int b = blockIdx.x;
    const int t = threadIdx.x;
    const int w = t >> 5;
    const int L = t & 31;
    const int base = 512 * w + 32 * (L & 15) + ((L < 16) ? 0 : 8);

    float cnt[16], rsd[16], ctr[16];
    v16h av;
#pragma unroll
    for (int j = 0; j < 16; ++j) {
        int idx = base + j + ((j >= 8) ? 8 : 0);
        float c0 = hcnt[(size_t)b * NBINS + idx];
        float r0 = hres[(size_t)b * NBINS + idx];
        cnt[j] = c0;
        rsd[j] = r0;
        ctr[j] = fmaf((float)idx + 0.5f, BINW, XMIN);
        av[j]  = (f16_t)c0;   // counts < 2048 -> exact in f16
    }

    __shared__ float red[8];
    __shared__ float fshared;
    float lo = -20.0f, hi = 20.0f;

    for (int iter = 0; iter < 40; ++iter) {
        float mid = 0.5f * (lo + hi);
        v16h bv;
        float corr = 0.0f;
#pragma unroll
        for (int j = 0; j < 16; ++j) {
            float s  = fast_sigmoid(ctr[j] + mid);
            f16_t h  = (f16_t)s;
            bv[j]    = h;
            float sh = (float)h;
            corr = fmaf(cnt[j], s - sh, corr);               // f16 rounding fixup
            corr = fmaf(rsd[j], s * (1.0f - s), corr);       // 1st-order Taylor
        }
        v8f acc = {};
        acc = __builtin_amdgcn_wmma_f32_16x16x32_f16(
                  false, av, false, bv, (short)0, acc, false, false);

        // extract diagonal D[m][m]: m<8 -> lane m, vgpr m; m>=8 -> lane m+16, vgpr m-8
        float dsel = 0.0f;
        {
            int  k  = (L < 8) ? L : (L - 24);
            bool on = (L < 8) || (L >= 24);
#pragma unroll
            for (int i = 0; i < 8; ++i)
                if (k == i) dsel = acc[i];
            if (!on) dsel = 0.0f;
        }
        float v = dsel + corr;
#pragma unroll
        for (int off = 16; off; off >>= 1) v += __shfl_xor(v, off, 32);
        if (L == 0) red[w] = v;
        __syncthreads();
        if (t == 0) {
            float tot = 0.0f;
#pragma unroll
            for (int i = 0; i < 8; ++i) tot += red[i];
            fshared = tot * (1.0f / (float)NPB) - 0.5f;
        }
        __syncthreads();
        float fm = fshared;
        if (fm < 0.0f) lo = mid; else hi = mid;
    }

    if (t == 0) {
        float c  = 0.5f * (lo + hi);
        cvals[b]    = c;
        out_tail[b] = c;
    }
}

// ---------------- 3) apply sigmoid(x + c_b), async-LDS staged stream ------
// grid = (128, BATCH); 8192 elements per block, 8 b128 iters per thread.
#define APPLY_ITERS 8
__global__ __launch_bounds__(256) void apply_kernel(const float* __restrict__ X,
                                                    const float* __restrict__ cvals,
                                                    float* __restrict__ out) {
    const int b  = blockIdx.y;
    const float cb = cvals[b];
    const size_t base = (size_t)b * NPB + (size_t)blockIdx.x * (NPB / 128);
    const f32x4* src = (const f32x4*)(X + base);
    f32x4*       dst = (f32x4*)(out + base);
    const int tid = threadIdx.x;

#ifdef HAS_ASYNC_LDS
    __shared__ f32x4 stage[2][256];
    // prime the 2-deep pipeline
    __builtin_amdgcn_global_load_async_to_lds_b128(
        to_global_v4(src + tid), to_lds_v4(&stage[0][tid]), 0, 0);
#pragma unroll
    for (int it = 0; it < APPLY_ITERS; ++it) {
        if (it + 1 < APPLY_ITERS) {
            __builtin_amdgcn_global_load_async_to_lds_b128(
                to_global_v4(src + (it + 1) * 256 + tid),
                to_lds_v4(&stage[(it + 1) & 1][tid]), 0, 0);
            __builtin_amdgcn_s_wait_asynccnt(1);
        } else {
            __builtin_amdgcn_s_wait_asynccnt(0);
        }
        asm volatile("" ::: "memory");
        f32x4 v = stage[it & 1][tid];
        f32x4 r;
        r.x = fast_sigmoid(v.x + cb);
        r.y = fast_sigmoid(v.y + cb);
        r.z = fast_sigmoid(v.z + cb);
        r.w = fast_sigmoid(v.w + cb);
        __builtin_nontemporal_store(r, &dst[it * 256 + tid]);
    }
#else
#pragma unroll
    for (int it = 0; it < APPLY_ITERS; ++it) {
        f32x4 v = src[it * 256 + tid];
        f32x4 r;
        r.x = fast_sigmoid(v.x + cb);
        r.y = fast_sigmoid(v.y + cb);
        r.z = fast_sigmoid(v.z + cb);
        r.w = fast_sigmoid(v.w + cb);
        __builtin_nontemporal_store(r, &dst[it * 256 + tid]);
    }
#endif
}

// ---------------- launch ----------------
extern "C" void kernel_launch(void* const* d_in, const int* in_sizes, int n_in,
                              void* d_out, int out_size, void* d_ws, size_t ws_size,
                              hipStream_t stream) {
    const float* X   = (const float*)d_in[0];
    float*       out = (float*)d_out;
    float*       ws  = (float*)d_ws;

    float* hcnt = ws;                       // [BATCH * NBINS]
    float* hres = ws + BATCH * NBINS;       // [BATCH * NBINS]
    float* cv   = ws + 2 * BATCH * NBINS;   // [BATCH]

    const int zn = 2 * BATCH * NBINS;
    zero_kernel<<<(zn + 255) / 256, 256, 0, stream>>>(ws, zn);

    hist_kernel<<<dim3(32, BATCH), 256, 0, stream>>>(X, hcnt, hres);

    bisect_kernel<<<BATCH, 256, 0, stream>>>(hcnt, hres, cv,
                                             out + (size_t)BATCH * NPB);

    apply_kernel<<<dim3(128, BATCH), 256, 0, stream>>>(X, cv, out);
}
